// LSTMLayerL2R_46514495816257
// MI455X (gfx1250) — matrix operand order repaired and checked by reading, not compiled, not run on hardware
//
#include <hip/hip_runtime.h>
#include <stdint.h>

#define T_STEPS 512
#define BATCH   64
#define ISZ     1024
#define HSZ     1024
#define G4H     4096

typedef __attribute__((ext_vector_type(16))) __bf16 v16bf;
typedef __attribute__((ext_vector_type(8)))  float  v8f;
typedef int v4i_vs __attribute__((vector_size(16)));   // gcc-style int4 (builtin param type)

#define AS1 __attribute__((address_space(1)))
#define AS3 __attribute__((address_space(3)))

union Frag16 {
    v16bf v;
    uint4 q[2];
};

__device__ __forceinline__ uint16_t f2bf(float f) {
    union { float f; uint32_t u; } v;
    v.f = f;
    uint32_t r = v.u + 0x7FFFu + ((v.u >> 16) & 1u);  // round-to-nearest-even
    return (uint16_t)(r >> 16);
}

__device__ __forceinline__ float sigmoidf_(float x) {
    return 1.0f / (1.0f + __expf(-x));
}

// ---- CDNA5 async global->LDS copy (per-lane 16B), ASYNCcnt tracked ----
__device__ __forceinline__ void async_copy16(const void* g, void* l) {
#if __has_builtin(__builtin_amdgcn_global_load_async_to_lds_b128)
    // param0: AS1 int4* (clang prints as "int __vector(4) __device__ *")
    __builtin_amdgcn_global_load_async_to_lds_b128(
        (AS1 v4i_vs*)(AS1 void*)g, (AS3 v4i_vs*)(AS3 void*)l, 0, 0);
#else
    asm volatile("global_load_async_to_lds_b128 %0, %1, off"
                 :: "v"((uint32_t)(uintptr_t)(AS3 void*)l), "v"(g)
                 : "memory");
#endif
}

__device__ __forceinline__ void wait_async0() {
#if __has_builtin(__builtin_amdgcn_s_wait_asynccnt)
    __builtin_amdgcn_s_wait_asynccnt(0);
#else
    asm volatile("s_wait_asynccnt 0x0" ::: "memory");
#endif
}

// A fragment: 16x32 bf16 tile, row-major source (stride ld elems).
// lane<16: M=lane, K in {k0..k0+7, k0+16..k0+23}; lane>=16: +8 K offset.
__device__ __forceinline__ void load_a_frag(Frag16& f, const uint16_t* base,
                                            int row0, int k0, int ld, int lane) {
    int row = row0 + (lane & 15);
    int kb  = k0 + 8 * (lane >> 4);
    const uint16_t* p = base + (size_t)row * ld + kb;
    f.q[0] = *(const uint4*)(p);
    f.q[1] = *(const uint4*)(p + 16);
}

// B fragment: 32x16 (KxN), B[k][n] = W[n0+n][k], W row-major [N][K] stride ld.
__device__ __forceinline__ void load_b_frag(Frag16& f, const uint16_t* base,
                                            int n0, int k0, int ld, int lane) {
    int col = n0 + (lane & 15);
    int kb  = k0 + 8 * (lane >> 4);
    const uint16_t* p = base + (size_t)col * ld + kb;
    f.q[0] = *(const uint4*)(p);
    f.q[1] = *(const uint4*)(p + 16);
}

// LDS-tile fragment loads; LDS rows padded to 40 elems (80B) for bank spread.
__device__ __forceinline__ void lds_a_frag(Frag16& f, const uint16_t* lA, int row, int lane) {
    const uint16_t* p = lA + row * 40 + 8 * (lane >> 4);
    f.q[0] = *(const uint4*)(p);
    f.q[1] = *(const uint4*)(p + 16);
}

__device__ __forceinline__ void lds_b_frag(Frag16& f, const uint16_t* lB, int row, int lane) {
    const uint16_t* p = lB + row * 40 + 8 * (lane >> 4);
    f.q[0] = *(const uint4*)(p);
    f.q[1] = *(const uint4*)(p + 16);
}

// ---------------- prep kernels ----------------

__global__ void cvt_bf16_kernel(const float* __restrict__ in, uint16_t* __restrict__ out, int n) {
    for (int i = blockIdx.x * blockDim.x + threadIdx.x; i < n; i += gridDim.x * blockDim.x)
        out[i] = f2bf(__builtin_nontemporal_load(in + i));
}

__global__ void bias_kernel(const float* __restrict__ a, const float* __restrict__ b,
                            float* __restrict__ o, int n) {
    for (int i = blockIdx.x * blockDim.x + threadIdx.x; i < n; i += gridDim.x * blockDim.x)
        o[i] = a[i] + b[i];
}

__global__ void init_state_kernel(const float* __restrict__ h0, const float* __restrict__ c0,
                                  uint16_t* __restrict__ hbf, float* __restrict__ cst, int n) {
    for (int i = blockIdx.x * blockDim.x + threadIdx.x; i < n; i += gridDim.x * blockDim.x) {
        hbf[i] = f2bf(h0[i]);
        cst[i] = c0[i];
    }
}

__global__ void finalize_kernel(const float* __restrict__ h_last, const float* __restrict__ cst,
                                float* __restrict__ hT, float* __restrict__ cT, int n) {
    for (int i = blockIdx.x * blockDim.x + threadIdx.x; i < n; i += gridDim.x * blockDim.x) {
        hT[i] = h_last[i];
        cT[i] = cst[i];
    }
}

// ---------------- phase 1: x_proj = x @ W_ih^T + (b_ih + b_hh) ----------------
// M=32768, N=4096, K=1024. WG tile 64(M) x 128(N), K-step 32, double-buffered
// async LDS staging: A tile 64x32 bf16, B tile 128x32 bf16 per buffer.
// 8 waves (2M x 4N), wave tile 32x32 = 2x2 WMMA C tiles.

// Issue one K-step tile: each thread contributes per-lane 16B async chunks.
__device__ __forceinline__ void async_tile_issue(const uint16_t* __restrict__ xbf,
                                                 const uint16_t* __restrict__ wih,
                                                 uint16_t* lA, uint16_t* lB,
                                                 int m0blk, int n0blk, int k0, int tid) {
    {   // A: 64 rows x 32 elems = 256 chunks of 16B (1 per thread)
        int row = tid >> 2, part = tid & 3;
        async_copy16(xbf + (size_t)(m0blk + row) * ISZ + k0 + part * 8,
                     lA + row * 40 + part * 8);
    }
#pragma unroll
    for (int j = 0; j < 2; ++j) {  // B: 128 rows x 32 elems = 512 chunks (2 per thread)
        int c = tid + 256 * j;
        int row = c >> 2, part = c & 3;
        async_copy16(wih + (size_t)(n0blk + row) * ISZ + k0 + part * 8,
                     lB + row * 40 + part * 8);
    }
}

__global__ __launch_bounds__(256) void gemm_xproj_kernel(const uint16_t* __restrict__ xbf,
                                                         const uint16_t* __restrict__ wih,
                                                         const float* __restrict__ bias,
                                                         float* __restrict__ xproj) {
    __shared__ uint16_t lA[2][64][40];    // 10 KB
    __shared__ uint16_t lB[2][128][40];   // 20 KB

    int tid  = threadIdx.x;
    int lane = tid & 31;
    int wid  = tid >> 5;
    int m0blk = blockIdx.y * 64;
    int n0blk = blockIdx.x * 128;
    int wm = (wid >> 2) * 32;  // wave row offset within tile
    int wn = (wid & 3) * 32;   // wave col offset within tile

    v8f acc[2][2] = {};

    async_tile_issue(xbf, wih, &lA[0][0][0], &lB[0][0][0], m0blk, n0blk, 0, tid);

    for (int kk = 0; kk < ISZ / 32; ++kk) {
        int buf = kk & 1;
        wait_async0();       // my 3 async copies for buf are done (in-order)
        __syncthreads();     // everyone's are done; prior reads of buf^1 finished
        if (kk + 1 < ISZ / 32)
            async_tile_issue(xbf, wih, &lA[buf ^ 1][0][0], &lB[buf ^ 1][0][0],
                             m0blk, n0blk, (kk + 1) * 32, tid);

        Frag16 a[2], b[2];
        lds_a_frag(a[0], &lA[buf][0][0], wm +      (lane & 15), lane);
        lds_a_frag(a[1], &lA[buf][0][0], wm + 16 + (lane & 15), lane);
        lds_b_frag(b[0], &lB[buf][0][0], wn +      (lane & 15), lane);
        lds_b_frag(b[1], &lB[buf][0][0], wn + 16 + (lane & 15), lane);
#pragma unroll
        for (int i = 0; i < 2; ++i)
#pragma unroll
            for (int j = 0; j < 2; ++j)
                acc[i][j] = __builtin_amdgcn_wmma_f32_16x16x32_bf16(
                    false, a[i].v, false, b[j].v, (short)0, acc[i][j], false, false);
    }

#pragma unroll
    for (int i = 0; i < 2; ++i)
#pragma unroll
        for (int j = 0; j < 2; ++j) {
            int col = n0blk + wn + 16 * j + (lane & 15);
            float bv = bias[col];
#pragma unroll
            for (int r = 0; r < 8; ++r) {
                int row = m0blk + wm + 16 * i + r + 8 * (lane >> 4);
                // x_proj is a write-once 512MB stream: keep it out of near caches
                __builtin_nontemporal_store(acc[i][j][r] + bv,
                                            &xproj[(size_t)row * G4H + col]);
            }
        }
}

// ---------------- phase 2: one LSTM timestep ----------------
// gates = x_proj[t] + h_prev @ W_hh^T; WG w owns h cols [16w,16w+16) and all 4
// gate groups for them (local fused nonlinearity via LDS). W_hh/h stay L2-hot.
__global__ __launch_bounds__(256) void lstm_step_kernel(const float* __restrict__ xproj_t,
                                                        const uint16_t* __restrict__ whh,
                                                        const uint16_t* __restrict__ h_in,
                                                        uint16_t* __restrict__ h_out,
                                                        float* __restrict__ c_st,
                                                        float* __restrict__ out_t) {
    __shared__ float gl[4][BATCH][16];  // 16 KB gate staging

    int lane = threadIdx.x & 31;
    int wid  = threadIdx.x >> 5;
    int j0   = blockIdx.x * 16;   // h-column base
    int mt   = wid >> 1;          // batch tile 0..3
    int gp   = wid & 1;           // gate pair: {i,f} or {g,o}
    int m0   = mt * 16;

    v8f acc[2] = {};
    for (int k0 = 0; k0 < HSZ; k0 += 32) {
        Frag16 a, b0, b1;
        load_a_frag(a, h_in, m0, k0, HSZ, lane);
        load_b_frag(b0, whh, (2 * gp)     * HSZ + j0, k0, HSZ, lane);
        load_b_frag(b1, whh, (2 * gp + 1) * HSZ + j0, k0, HSZ, lane);
        acc[0] = __builtin_amdgcn_wmma_f32_16x16x32_bf16(false, a.v, false, b0.v,
                                                         (short)0, acc[0], false, false);
        acc[1] = __builtin_amdgcn_wmma_f32_16x16x32_bf16(false, a.v, false, b1.v,
                                                         (short)0, acc[1], false, false);
    }

    int colL = lane & 15;
#pragma unroll
    for (int g = 0; g < 2; ++g) {
        int gate = 2 * gp + g;
        int gc   = gate * HSZ + j0 + colL;
#pragma unroll
        for (int r = 0; r < 8; ++r) {
            int row = m0 + r + 8 * (lane >> 4);
            // x_proj slice is read exactly once: non-temporal
            gl[gate][row][colL] = acc[g][r] +
                __builtin_nontemporal_load(&xproj_t[(size_t)row * G4H + gc]);
        }
    }
    __syncthreads();

#pragma unroll
    for (int rr = 0; rr < 4; ++rr) {
        int f    = threadIdx.x + 256 * rr;  // 0..1023 over (64 rows x 16 cols)
        int brow = f >> 4;
        int col  = f & 15;
        float ig = sigmoidf_(gl[0][brow][col]);
        float fg = sigmoidf_(gl[1][brow][col]);
        float gg = tanhf(gl[2][brow][col]);
        float og = sigmoidf_(gl[3][brow][col]);
        int idx  = brow * HSZ + j0 + col;
        float cn = fg * c_st[idx] + ig * gg;
        c_st[idx] = cn;
        float hn  = og * tanhf(cn);
        out_t[idx] = hn;
        h_out[idx] = f2bf(hn);
    }
}

// ---------------- host ----------------

extern "C" void kernel_launch(void* const* d_in, const int* in_sizes, int n_in,
                              void* d_out, int out_size, void* d_ws, size_t ws_size,
                              hipStream_t stream) {
    const float* x    = (const float*)d_in[0];
    const float* h0   = (const float*)d_in[1];
    const float* c0   = (const float*)d_in[2];
    const float* W_ih = (const float*)d_in[3];
    const float* W_hh = (const float*)d_in[4];
    const float* b_ih = (const float*)d_in[5];
    const float* b_hh = (const float*)d_in[6];
    float* out = (float*)d_out;
    char*  ws  = (char*)d_ws;

    // workspace layout (bytes)
    const size_t XBF_B  = (size_t)T_STEPS * BATCH * ISZ * 2;  // 64 MB
    const size_t W_B    = (size_t)G4H * ISZ * 2;              // 8 MB each
    const size_t BIAS_B = (size_t)G4H * 4;
    const size_t HBF_B  = (size_t)BATCH * HSZ * 2;
    const size_t CST_B  = (size_t)BATCH * HSZ * 4;

    uint16_t* x_bf   = (uint16_t*)(ws);
    uint16_t* wih_bf = (uint16_t*)(ws + XBF_B);
    uint16_t* whh_bf = (uint16_t*)(ws + XBF_B + W_B);
    float*    bias   = (float*)   (ws + XBF_B + 2 * W_B);
    uint16_t* hbf0   = (uint16_t*)(ws + XBF_B + 2 * W_B + BIAS_B);
    uint16_t* hbf1   = (uint16_t*)(ws + XBF_B + 2 * W_B + BIAS_B + HBF_B);
    float*    c_st   = (float*)   (ws + XBF_B + 2 * W_B + BIAS_B + 2 * HBF_B);
    float*    xproj  = (float*)   (ws + XBF_B + 2 * W_B + BIAS_B + 2 * HBF_B + CST_B);

    // prep: bf16 conversions, bias sum, state init
    cvt_bf16_kernel<<<8192, 256, 0, stream>>>(x, x_bf, T_STEPS * BATCH * ISZ);
    cvt_bf16_kernel<<<4096, 256, 0, stream>>>(W_ih, wih_bf, G4H * ISZ);
    cvt_bf16_kernel<<<4096, 256, 0, stream>>>(W_hh, whh_bf, G4H * HSZ);
    bias_kernel<<<16, 256, 0, stream>>>(b_ih, b_hh, bias, G4H);
    init_state_kernel<<<256, 256, 0, stream>>>(h0, c0, hbf0, c_st, BATCH * HSZ);

    // phase 1: big GEMM for all timesteps (async-LDS staged WMMA)
    gemm_xproj_kernel<<<dim3(G4H / 128, (T_STEPS * BATCH) / 64), 256, 0, stream>>>(
        x_bf, wih_bf, bias, xproj);

    // phase 2: sequential recurrence (ping-pong bf16 h buffers)
    for (int t = 0; t < T_STEPS; ++t) {
        const uint16_t* hin  = (t & 1) ? hbf1 : hbf0;
        uint16_t*       hout = (t & 1) ? hbf0 : hbf1;
        lstm_step_kernel<<<HSZ / 16, 256, 0, stream>>>(
            xproj + (size_t)t * BATCH * G4H, whh_bf, hin, hout, c_st,
            out + (size_t)t * BATCH * HSZ);
    }

    // tail: (h_T, c_T)
    float* hT = out + (size_t)T_STEPS * BATCH * HSZ;
    float* cT = hT + (size_t)BATCH * HSZ;
    finalize_kernel<<<256, 256, 0, stream>>>(
        out + (size_t)(T_STEPS - 1) * BATCH * HSZ, c_st, hT, cT, BATCH * HSZ);
}
